// Separate2_58875411693695
// MI455X (gfx1250) — matrix operand order repaired and checked by reading, not compiled, not run on hardware
//
#include <hip/hip_runtime.h>
#include <hip/hip_bf16.h>

typedef __attribute__((ext_vector_type(16))) _Float16 v16h;
typedef __attribute__((ext_vector_type(8)))  _Float16 v8h;
typedef __attribute__((ext_vector_type(8)))  float    v8f;

#define KTAG 48   // K speakers
#define DDIM 256  // embedding dim (reduction)
#define PDIM 40   // phonemes
#define PPAD 48   // padded to 3 x 16 WMMA tiles
#define CW   32   // gate hidden width

// ---------------------------------------------------------------------------
// Kernel 1: L2-normalize along D and pack to f16 in WMMA-friendly layout:
//   dst[(k*PPAD + p)*DDIM + d] = src[k,d,p] / max(||src[k,:,p]||, 1e-12)
// Rows p in [40,48) are zero (padding) so GEMM tiles need no bounds checks.
// grid = (PPAD, K, 2), block = 256 (one thread per d)
// ---------------------------------------------------------------------------
__global__ void __launch_bounds__(256)
normalize_pack(const float* __restrict__ enroll,
               const float* __restrict__ test,
               _Float16* __restrict__ Anorm,
               _Float16* __restrict__ Bnorm) {
  const int p = blockIdx.x;
  const int k = blockIdx.y;
  const int d = threadIdx.x;
  const float* src = blockIdx.z ? test : enroll;
  _Float16*   dst = blockIdx.z ? Bnorm : Anorm;

  float x = 0.0f;
  if (p < PDIM) x = src[(k * DDIM + d) * PDIM + p];

  __shared__ float red[256];
  red[d] = x * x;
  __syncthreads();
  #pragma unroll
  for (int s = 128; s > 0; s >>= 1) {
    if (d < s) red[d] += red[d + s];
    __syncthreads();
  }
  const float scale = 1.0f / fmaxf(sqrtf(red[0]), 1e-12f);
  dst[(size_t)(k * PPAD + p) * DDIM + d] = (_Float16)(x * scale);
}

// ---------------------------------------------------------------------------
// Kernel 2: one block per (k,l) pair. 9 waves; wave w computes the 16x16 tile
// (ti,tj) of the 48x48 padded pho tile via v_wmma_f32_16x16x32_f16 over D=256,
// dumps it to LDS, then the block runs the fused gate/tanh epilogue.
// ---------------------------------------------------------------------------
__global__ void __launch_bounds__(288)
pair_kernel(const _Float16* __restrict__ Anorm,
            const _Float16* __restrict__ Bnorm,
            const float* __restrict__ fc1_w, const float* __restrict__ fc1_b,
            const float* __restrict__ fc2_w,
            const float* __restrict__ fc3_w, const float* __restrict__ fc3_b,
            const float* __restrict__ fc4_w,
            const float* __restrict__ fc5_w,
            float* __restrict__ out) {
  const int pair = blockIdx.x;
  const int k = pair / KTAG;   // enroll speaker
  const int l = pair % KTAG;   // test speaker
  const int tid  = threadIdx.x;
  const int wave = tid >> 5;
  const int lane = tid & 31;

  __shared__ float s_pho[PPAD * 49];          // 48x48 tile, stride 49 (bank-friendly)
  __shared__ float s_w1[CW], s_b1[CW], s_w2[CW];
  __shared__ float s_w3[CW], s_b3[CW], s_w4[CW];
  __shared__ float s_rowsum[PDIM], s_rowcnt[PDIM];
  __shared__ float s_y2[PDIM], s_c2[PDIM], s_ys[PDIM], s_cs[PDIM];

  if (tid < CW) {
    s_w1[tid] = fc1_w[tid]; s_b1[tid] = fc1_b[tid]; s_w2[tid] = fc2_w[tid];
    s_w3[tid] = fc3_w[tid]; s_b3[tid] = fc3_b[tid]; s_w4[tid] = fc4_w[tid];
  }
  if (tid < PDIM) { s_rowsum[tid] = 0.0f; s_rowcnt[tid] = 0.0f; }

  // ---------------- WMMA GEMM: pho[p,q] = sum_d e[k,p,d] * t[l,q,d] ----------
  const int ti = wave / 3;                 // M tile (enroll phoneme)
  const int tj = wave % 3;                 // N tile (test phoneme)
  const int l15 = lane & 15;
  const int hi  = lane >> 4;               // upper lane half?
  // A fragment: lane holds row M = ti*16 + (lane&15); K chunks at
  //   base = hi*8 and base+16 (8 contiguous halfs each)   [ISA 16-bit A layout]
  const _Float16* rowA = Anorm + (size_t)(k * PPAD + ti * 16 + l15) * DDIM;
  // B fragment: lane holds col N = tj*16 + (lane&15); K chunks at
  //   base = hi*8 and base+16 (8 contiguous halfs each)   [ISA B layout]
  const _Float16* rowB = Bnorm + (size_t)(l * PPAD + tj * 16 + l15) * DDIM;

  v8f acc = {};
  #pragma unroll
  for (int kk = 0; kk < DDIM; kk += 32) {
    const int ka = kk + hi * 8;
    const int kb = kk + hi * 8;
    v8h a0 = *(const v8h*)(rowA + ka);
    v8h a1 = *(const v8h*)(rowA + ka + 16);
    v8h b0 = *(const v8h*)(rowB + kb);
    v8h b1 = *(const v8h*)(rowB + kb + 16);
    v16h a, b;
    #pragma unroll
    for (int i = 0; i < 8; i++) {
      a[i] = a0[i]; a[8 + i] = a1[i];
      b[i] = b0[i]; b[8 + i] = b1[i];
    }
    // D = A(16x32) * B(32x16) + C, f32 accumulate
    acc = __builtin_amdgcn_wmma_f32_16x16x32_f16(
        /*neg_a=*/false, a, /*neg_b=*/false, b,
        /*c_mod=*/(short)0, acc, /*reuse_a=*/false, /*reuse_b=*/false);
  }

  // C/D layout: VGPR r -> M = r (lanes 0-15) / r+8 (lanes 16-31), N = lane&15
  {
    const int n     = tj * 16 + l15;
    const int mbase = ti * 16 + hi * 8;
    #pragma unroll
    for (int r = 0; r < 8; r++) s_pho[(mbase + r) * 49 + n] = acc[r];
  }
  __syncthreads();

  // ---------------- epilogue stage 1: gate every pho element ----------------
  for (int i = tid; i < PDIM * PDIM; i += 288) {
    const int p = i / PDIM;
    const int q = i - p * PDIM;
    const float x = s_pho[p * 49 + q];
    float w = 0.0f;
    #pragma unroll
    for (int c = 0; c < CW; c++) w += s_w4[c] * tanhf(x * s_w3[c] + s_b3[c]);
    atomicAdd(&s_rowsum[p], tanhf(w * x));
    if (x != 0.0f) atomicAdd(&s_rowcnt[p], 1.0f);
  }
  __syncthreads();

  // ---------------- stage 2 per row + same-phoneme diagonal -----------------
  if (tid < PDIM) {
    const int p = tid;
    const float m = s_rowsum[p] / (s_rowcnt[p] + 1e-6f);
    float w2 = 0.0f;
    #pragma unroll
    for (int c = 0; c < CW; c++) w2 += s_w4[c] * tanhf(m * s_w3[c] + s_b3[c]);
    s_y2[p] = tanhf(w2 * m);
    s_c2[p] = (m != 0.0f) ? 1.0f : 0.0f;

    const float xd = s_pho[p * 49 + p];
    float sw = 0.0f;
    #pragma unroll
    for (int c = 0; c < CW; c++) sw += s_w2[c] * tanhf(xd * s_w1[c] + s_b1[c]);
    s_ys[p] = tanhf(sw * xd);
    s_cs[p] = (xd != 0.0f) ? 1.0f : 0.0f;
  }
  __syncthreads();

  if (tid == 0) {
    float sum2 = 0.f, c2 = 0.f, ss = 0.f, cs = 0.f;
    #pragma unroll
    for (int p = 0; p < PDIM; p++) {
      sum2 += s_y2[p]; c2 += s_c2[p];
      ss   += s_ys[p]; cs += s_cs[p];
    }
    const float diff_score = sum2 / (c2 + 1e-6f);
    const float same_score = ss   / (cs + 1e-6f);
    out[pair] = tanhf(same_score * fc5_w[0] + diff_score * fc5_w[1]);
  }
}

// ---------------------------------------------------------------------------
extern "C" void kernel_launch(void* const* d_in, const int* in_sizes, int n_in,
                              void* d_out, int out_size, void* d_ws, size_t ws_size,
                              hipStream_t stream) {
  (void)in_sizes; (void)n_in; (void)out_size; (void)ws_size;
  const float* enroll = (const float*)d_in[0];
  const float* test   = (const float*)d_in[1];
  const float* fc1_w  = (const float*)d_in[2];
  const float* fc1_b  = (const float*)d_in[3];
  const float* fc2_w  = (const float*)d_in[4];
  const float* fc3_w  = (const float*)d_in[5];
  const float* fc3_b  = (const float*)d_in[6];
  const float* fc4_w  = (const float*)d_in[7];
  const float* fc5_w  = (const float*)d_in[8];
  float* out = (float*)d_out;

  _Float16* Anorm = (_Float16*)d_ws;                               // 1.18 MB
  _Float16* Bnorm = Anorm + (size_t)KTAG * PPAD * DDIM;            // 1.18 MB

  dim3 gridN(PPAD, KTAG, 2);
  normalize_pack<<<gridN, 256, 0, stream>>>(enroll, test, Anorm, Bnorm);

  pair_kernel<<<KTAG * KTAG, 288, 0, stream>>>(
      Anorm, Bnorm, fc1_w, fc1_b, fc2_w, fc3_w, fc3_b, fc4_w, fc5_w, out);
}
